// CasualAttention_3607772528983
// MI455X (gfx1250) — matrix operand order repaired and checked by reading, not compiled, not run on hardware
//
#include <hip/hip_runtime.h>
#include <hip/hip_bf16.h>

#define BATCH 4
#define SEQ   2048
#define DIM   1024

typedef __bf16 bf16_t;
typedef __attribute__((ext_vector_type(16))) __bf16 v16bf;
typedef __attribute__((ext_vector_type(8)))  float  v8f;

static __device__ inline v8f wmma_bf16(v16bf a, v16bf b, v8f c) {
  // D = A(16x32 bf16) * B(32x16 bf16) + C(16x16 f32)
  return __builtin_amdgcn_wmma_f32_16x16x32_bf16(false, a, false, b,
                                                 (short)0, c, false, false);
}

// ---- fragment loaders, layouts per CDNA5 ISA 7.12.2 ----
// A-matrix 16x32 bf16: lane = row M (0..15 in both lane halves),
//   lanes 0-15:  v16 elems 0..7 -> K=0..7,  elems 8..15 -> K=16..23
//   lanes 16-31: v16 elems 0..7 -> K=8..15, elems 8..15 -> K=24..31
// rp points at this lane's row, at the K-base.
static __device__ inline v16bf afrag_bf16(const bf16_t* rp, int half) {
  v16bf a;
#pragma unroll
  for (int i = 0; i < 8; ++i) {
    a[i]     = rp[half * 8 + i];
    a[8 + i] = rp[16 + half * 8 + i];
  }
  return a;
}
// B-matrix 32x16 bf16: lane = column N (0..15), same K order as A:
// v16 elems 0..7 -> K = half*8 + e, elems 8..15 -> K = 16 + half*8 + (e-8).
// rp points at this lane's column-source row, at the K-base.
static __device__ inline v16bf bfrag_bf16(const bf16_t* rp, int half) {
  v16bf b;
#pragma unroll
  for (int i = 0; i < 8; ++i) {
    b[i]     = rp[half * 8 + i];
    b[8 + i] = rp[16 + half * 8 + i];
  }
  return b;
}

// ---------------------------------------------------------------------------
// Kernel 0: one-shot f32 -> bf16 conversion (4 elements/thread, n % 4 == 0).
// ---------------------------------------------------------------------------
__global__ void __launch_bounds__(256)
cvt_f32_bf16_kernel(const float* __restrict__ src, bf16_t* __restrict__ dst,
                    int n4) {
  int i = blockIdx.x * blockDim.x + threadIdx.x;
  if (i < n4) {
    float4 v = ((const float4*)src)[i];
    bf16_t* d = dst + (size_t)i * 4;
    d[0] = (bf16_t)v.x; d[1] = (bf16_t)v.y;
    d[2] = (bf16_t)v.z; d[3] = (bf16_t)v.w;
  }
}

// ---------------------------------------------------------------------------
// Kernel 1: fused QKV projection on pre-converted bf16.  C = x @ W^T.
// grid = (BATCH*SEQ/16, DIM/64, 3), block = 32.
// One wave computes a 16x64 tile with an explicit 1-deep software pipeline:
// the next k-step's A-frag + 4 B-frags are loaded into fresh registers before
// the current 4 WMMAs issue, so loads overlap matrix ops instead of the
// s_wait_loadcnt 0x0 full-serialization seen without pipelining.
// Q,K stored bf16 row-major [B][N][D]; V stored TRANSPOSED bf16 [B][D][N].
// ---------------------------------------------------------------------------
__global__ void __launch_bounds__(32)
qkv_gemm_kernel(const bf16_t* __restrict__ Xb,
                const bf16_t* __restrict__ Wb,   // [3][DIM][DIM] bf16
                bf16_t* __restrict__ Qb,
                bf16_t* __restrict__ Kb,
                bf16_t* __restrict__ Vt) {
  const int lane = threadIdx.x & 31;
  const int half = lane >> 4;
  const int n16  = lane & 15;
  const int tm   = blockIdx.x;                 // 16-row tile over BATCH*SEQ
  const int tn   = blockIdx.y;                 // 64-col tile over DIM
  const int z    = blockIdx.z;                 // 0:Q 1:K 2:V

  const bf16_t* W    = Wb + (size_t)z * DIM * DIM;
  const bf16_t* xrow = Xb + (size_t)(tm * 16 + n16) * DIM;
  const bf16_t* wrow = W  + (size_t)(tn * 64 + n16) * DIM;

  v8f acc[4] = {};

  // pipeline prologue: k = 0 fragments
  v16bf a = afrag_bf16(xrow, half);
  v16bf b0 = bfrag_bf16(wrow,                        half);
  v16bf b1 = bfrag_bf16(wrow + (size_t)16  * DIM,    half);
  v16bf b2 = bfrag_bf16(wrow + (size_t)32  * DIM,    half);
  v16bf b3 = bfrag_bf16(wrow + (size_t)48  * DIM,    half);

#pragma unroll 2
  for (int k = 0; k < DIM - 32; k += 32) {
    // prefetch k+32 into fresh registers (loads overlap the WMMAs below)
    v16bf an  = afrag_bf16(xrow + k + 32, half);
    v16bf bn0 = bfrag_bf16(wrow + k + 32,                     half);
    v16bf bn1 = bfrag_bf16(wrow + (size_t)16 * DIM + k + 32,  half);
    v16bf bn2 = bfrag_bf16(wrow + (size_t)32 * DIM + k + 32,  half);
    v16bf bn3 = bfrag_bf16(wrow + (size_t)48 * DIM + k + 32,  half);

    acc[0] = wmma_bf16(a, b0, acc[0]);
    acc[1] = wmma_bf16(a, b1, acc[1]);
    acc[2] = wmma_bf16(a, b2, acc[2]);
    acc[3] = wmma_bf16(a, b3, acc[3]);

    a = an; b0 = bn0; b1 = bn1; b2 = bn2; b3 = bn3;
  }
  // epilogue
  acc[0] = wmma_bf16(a, b0, acc[0]);
  acc[1] = wmma_bf16(a, b1, acc[1]);
  acc[2] = wmma_bf16(a, b2, acc[2]);
  acc[3] = wmma_bf16(a, b3, acc[3]);

  // C layout: lane L, vgpr r -> M = half*8 + r, N = L&15
  if (z < 2) {
    bf16_t* dst = z ? Kb : Qb;
#pragma unroll
    for (int t = 0; t < 4; ++t) {
      const int col = tn * 64 + t * 16 + n16;
#pragma unroll
      for (int r = 0; r < 8; ++r) {
        int gr = tm * 16 + half * 8 + r;
        dst[(size_t)gr * DIM + col] = (bf16_t)acc[t][r];
      }
    }
  } else {
#pragma unroll
    for (int t = 0; t < 4; ++t) {
      const int col = tn * 64 + t * 16 + n16;
#pragma unroll
      for (int r = 0; r < 8; ++r) {
        int gr = tm * 16 + half * 8 + r;
        int bb = gr >> 11;                     // / SEQ
        int nn = gr & (SEQ - 1);
        Vt[((size_t)bb * DIM + col) * SEQ + nn] = (bf16_t)acc[t][r];
      }
    }
  }
}

// ---------------------------------------------------------------------------
// Kernel 2: flash attention over bf16 Q,K,Vt with f32 accumulation.
// grid = (SEQ/16, BATCH), block = 256 (8 waves).
// Block owns 16 query rows; wave w owns output columns [w*128, w*128+128).
// Key tiles are 32 wide; S(16x32) reduced across waves via LDS ds_add_f32.
// ---------------------------------------------------------------------------
__global__ void __launch_bounds__(256)
flash_attn_kernel(const bf16_t* __restrict__ Qb,
                  const bf16_t* __restrict__ Kb,
                  const bf16_t* __restrict__ Vt,
                  float* __restrict__ out) {
  __shared__ float  sS[16 * 32];
  __shared__ bf16_t sP[16 * 32];
  __shared__ float  sAlpha[16];
  __shared__ float  sInvL[16];

  const int tid  = threadIdx.x;
  const int w    = tid >> 5;
  const int lane = tid & 31;
  const int half = lane >> 4;
  const int n16  = lane & 15;
  const int q0   = blockIdx.x * 16;
  const int b    = blockIdx.y;

  const float scale = 0.03125f;                // 1/sqrt(1024)

  v8f acc[8] = {};                             // 16 x 128 f32 output slice
  float m_i = -3.0e38f, l_i = 0.0f;            // live only in tid<16

  const bf16_t* qrow = Qb + (size_t)(b * SEQ + q0 + n16) * DIM;
  const int jmax = (q0 + 15) >> 5;

  for (int j = 0; j <= jmax; ++j) {
    // zero the 16x32 score reduction buffer
    sS[tid] = 0.0f;
    sS[tid + 256] = 0.0f;
    __syncthreads();

    // partial S = Q * K^T over this wave's 128-wide slice of D
    v8f s0 = {}, s1 = {};
#pragma unroll
    for (int kk = 0; kk < 4; ++kk) {
      const int d0 = w * 128 + kk * 32;
      v16bf a = afrag_bf16(qrow + d0, half);
      const bf16_t* k0 = Kb + (size_t)(b * SEQ + j * 32 + n16) * DIM + d0;
      v16bf bk0 = bfrag_bf16(k0, half);
      v16bf bk1 = bfrag_bf16(k0 + (size_t)16 * DIM, half);
      s0 = wmma_bf16(a, bk0, s0);
      s1 = wmma_bf16(a, bk1, s1);
    }
#pragma unroll
    for (int r = 0; r < 8; ++r) {
      const int row = half * 8 + r;
      atomicAdd(&sS[row * 32 + n16],      s0[r]);
      atomicAdd(&sS[row * 32 + 16 + n16], s1[r]);
    }
    __syncthreads();

    // online softmax: one thread per query row
    if (tid < 16) {
      const int qrow_g = q0 + tid;
      float mloc = -3.0e38f;
#pragma unroll 8
      for (int c = 0; c < 32; ++c) {
        float s = sS[tid * 32 + c] * scale;
        if (j * 32 + c > qrow_g) s = -3.0e38f; // causal mask
        sS[tid * 32 + c] = s;
        mloc = fmaxf(mloc, s);
      }
      const float mnew  = fmaxf(m_i, mloc);
      const float alpha = __expf(m_i - mnew);
      float psum = 0.0f;
#pragma unroll 8
      for (int c = 0; c < 32; ++c) {
        float p = __expf(sS[tid * 32 + c] - mnew);
        psum += p;
        sP[tid * 32 + c] = (bf16_t)p;
      }
      l_i = l_i * alpha + psum;
      m_i = mnew;
      sAlpha[tid] = alpha;
    }
    __syncthreads();

    // rescale running output and accumulate P(16x32) * V(32x128-slice)
    v16bf ap = afrag_bf16(&sP[n16 * 32], half);
#pragma unroll
    for (int t = 0; t < 8; ++t) {
      const int col = w * 128 + t * 16 + n16;
      const bf16_t* vrp = Vt + ((size_t)b * DIM + col) * SEQ + j * 32;
      v16bf bv = bfrag_bf16(vrp, half);
#pragma unroll
      for (int r = 0; r < 8; ++r) acc[t][r] *= sAlpha[half * 8 + r];
      acc[t] = wmma_bf16(ap, bv, acc[t]);
    }
    // next iteration has two barriers before sP/sAlpha are rewritten; sS is free after the barrier above.
  }

  if (tid < 16) sInvL[tid] = 1.0f / l_i;
  __syncthreads();

#pragma unroll
  for (int t = 0; t < 8; ++t) {
    const int col = w * 128 + t * 16 + n16;
#pragma unroll
    for (int r = 0; r < 8; ++r) {
      const int row = q0 + half * 8 + r;
      out[(size_t)(b * SEQ + row) * DIM + col] = acc[t][r] * sInvL[half * 8 + r];
    }
  }
}

// ---------------------------------------------------------------------------
extern "C" void kernel_launch(void* const* d_in, const int* in_sizes, int n_in,
                              void* d_out, int out_size, void* d_ws, size_t ws_size,
                              hipStream_t stream) {
  (void)in_sizes; (void)n_in; (void)out_size; (void)ws_size;
  const float* x  = (const float*)d_in[0];
  const float* Wq = (const float*)d_in[1];
  const float* Wk = (const float*)d_in[2];
  const float* Wv = (const float*)d_in[3];
  float* out = (float*)d_out;

  // workspace layout (bf16):
  //   Xb [B*N][D] 16MB | Wb [3][D][D] 6MB | Qb 16MB | Kb 16MB | Vt [B][D][N] 16MB
  const size_t XN = (size_t)BATCH * SEQ * DIM;   // 8.4M elems
  const size_t WN = (size_t)DIM * DIM;           // 1M elems per matrix
  bf16_t* Xb = (bf16_t*)d_ws;
  bf16_t* Wb = Xb + XN;
  bf16_t* Qb = Wb + 3 * WN;
  bf16_t* Kb = Qb + XN;
  bf16_t* Vt = Kb + XN;

  // one-shot conversions
  {
    int n4 = (int)(XN / 4);
    cvt_f32_bf16_kernel<<<(n4 + 255) / 256, 256, 0, stream>>>(x, Xb, n4);
    int w4 = (int)(WN / 4);
    cvt_f32_bf16_kernel<<<(w4 + 255) / 256, 256, 0, stream>>>(Wq, Wb, w4);
    cvt_f32_bf16_kernel<<<(w4 + 255) / 256, 256, 0, stream>>>(Wk, Wb + WN, w4);
    cvt_f32_bf16_kernel<<<(w4 + 255) / 256, 256, 0, stream>>>(Wv, Wb + 2 * WN, w4);
  }

  dim3 g1(BATCH * SEQ / 16, DIM / 64, 3);
  qkv_gemm_kernel<<<g1, 32, 0, stream>>>(Xb, Wb, Qb, Kb, Vt);

  dim3 g2(SEQ / 16, BATCH);
  flash_attn_kernel<<<g2, 256, 0, stream>>>(Qb, Kb, Vt, out);
}
